// VectorQuantizer_56607668961486
// MI455X (gfx1250) — matrix operand order repaired and checked by reading, not compiled, not run on hardware
//
#include <hip/hip_runtime.h>

typedef __attribute__((ext_vector_type(16))) __bf16 v16bf;
typedef __attribute__((ext_vector_type(8)))  __bf16 v8bf;
typedef __attribute__((ext_vector_type(8)))  float  v8f;

#define DDIM     256
#define KCODES   1024
#define HW       4096
#define MTILE    256                 // rows per block (8 waves x 32 rows)
#define PITCH    264                 // ushorts per LDS row = 528B = 33*16B (bank-spread, 16B aligned)
#define CCHUNK   128                 // codebook codes staged per chunk
#define NCHUNKS  (KCODES / CCHUNK)   // 8
#define NZ       33554432u           // 32*256*64*64
#define NIDX     131072u             // 32*64*64
#define LOSS_OFF (NZ + NIDX)
#define INV_N    (1.25f / 33554432.f)

__device__ __forceinline__ unsigned short f2bf(float f) {
  unsigned int u = __float_as_uint(f);
  u += 0x7FFFu + ((u >> 16) & 1u);       // round-to-nearest-even
  return (unsigned short)(u >> 16);
}

union Frag { v16bf v; v8bf h[2]; };

// ---- prep: e_sq[k] = ||codebook[k]||^2  (f32, exact); zero the loss slot ----
__global__ void vq_prep(const float* __restrict__ cb, float* __restrict__ esq,
                        float* __restrict__ out_loss) {
  int k = blockIdx.x * blockDim.x + threadIdx.x;
  if (k < KCODES) {
    const float4* row = (const float4*)(cb + (size_t)k * DDIM);
    float s = 0.f;
#pragma unroll 4
    for (int j = 0; j < DDIM / 4; ++j) {
      float4 v = row[j];
      s += v.x * v.x + v.y * v.y + v.z * v.z + v.w * v.w;
    }
    esq[k] = s;
  }
  if (k == 0) *out_loss = 0.f;
}

// ---- main: fused distance-GEMM (bf16 WMMA, 2 M-tiles/wave) + argmin + gather + loss ----
__global__ void __launch_bounds__(256, 1)
vq_main(const float* __restrict__ z, const float* __restrict__ cb,
        const float* __restrict__ esq_g, float* __restrict__ out) {
  extern __shared__ unsigned char smem[];
  unsigned short* zA    = (unsigned short*)smem;            // 256 * 264 ushorts
  unsigned short* cbS   = zA + MTILE * PITCH;               // 128 * 264 ushorts
  float*          esqS  = (float*)(cbS + CCHUNK * PITCH);   // 128
  float*          zsqS  = esqS + CCHUNK;                    // 256
  int*            idxS  = (int*)(zsqS + MTILE);             // 256
  float*          lossS = (float*)(idxS + MTILE);           // 1

  const int tid  = threadIdx.x;
  const int lane = tid & 31;
  const int wave = tid >> 5;
  const int lo16 = lane & 15;
  const int hi   = lane >> 4;

  const int n0  = blockIdx.x * MTILE;
  const int b   = n0 >> 12;            // / 4096  (256 | 4096, tile never crosses batch)
  const int hw0 = n0 & 4095;
  const float* zBase = z + (size_t)b * ((size_t)DDIM * HW) + hw0;

  if (tid == 0) lossS[0] = 0.f;

  // phase 1: stage z tile f32->bf16 (one thread per row; coalesced over hw),
  //          accumulate exact ||z||^2 per row (no atomics needed)
  {
    const int r = tid;                       // 256 threads == 256 rows
    float zs = 0.f;
#pragma unroll 4
    for (int d = 0; d < DDIM; ++d) {
      float v = zBase[(size_t)d * HW + r];
      zA[r * PITCH + d] = f2bf(v);
      zs += v * v;
    }
    zsqS[r] = zs;
  }
  __syncthreads();

  // preload this wave's TWO 16-row A fragment sets (rows rowBase..rowBase+31)
  // 16-bit A 16x32 layout: lane(l&15)=M; lanes0-15 K=0-7,16-23; lanes16-31 K=8-15,24-31
  const int rowBase = wave * 32;
  const int arow0   = rowBase + lo16;
  const int arow1   = arow0 + 16;
  Frag afrag0[8], afrag1[8];
#pragma unroll
  for (int dk = 0; dk < 8; ++dk) {
    const unsigned short* ap0 = zA + arow0 * PITCH + 32 * dk + 8 * hi;
    const unsigned short* ap1 = zA + arow1 * PITCH + 32 * dk + 8 * hi;
    afrag0[dk].h[0] = *(const v8bf*)(ap0);
    afrag0[dk].h[1] = *(const v8bf*)(ap0 + 16);
    afrag1[dk].h[0] = *(const v8bf*)(ap1);
    afrag1[dk].h[1] = *(const v8bf*)(ap1 + 16);
  }

  float minv0[8], minv1[8]; int mini0[8], mini1[8];
#pragma unroll
  for (int i = 0; i < 8; ++i) {
    minv0[i] = 3.4e38f; mini0[i] = 0;
    minv1[i] = 3.4e38f; mini1[i] = 0;
  }

  for (int c = 0; c < NCHUNKS; ++c) {
    __syncthreads();                         // protect cbS reuse
    {
      const int cbase = c * CCHUNK;
      if (tid < CCHUNK) esqS[tid] = esq_g[cbase + tid];
      const float* src = cb + (size_t)cbase * DDIM;
#pragma unroll 4
      for (int l = tid; l < CCHUNK * DDIM; l += 256) {
        int code = l >> 8;
        int d    = l & 255;
        cbS[code * PITCH + d] = f2bf(src[l]);
      }
      if (c + 1 < NCHUNKS)
        __builtin_prefetch(cb + (size_t)(cbase + CCHUNK) * DDIM, 0, 1);
    }
    __syncthreads();

#pragma unroll 1
    for (int kt = 0; kt < CCHUNK / 16; ++kt) {   // 8 column tiles of 16 codes
      const int codeLocal = kt * 16 + lo16;      // this lane's column N
      const unsigned short* bp = cbS + codeLocal * PITCH + 8 * hi;
      v8f acc0 = {0.f, 0.f, 0.f, 0.f, 0.f, 0.f, 0.f, 0.f};
      v8f acc1 = {0.f, 0.f, 0.f, 0.f, 0.f, 0.f, 0.f, 0.f};
#pragma unroll
      for (int dk = 0; dk < 8; ++dk) {           // K = 256 in steps of 32
        Frag bfrag;                              // B 32x16: lanes0-15 K=0-7,16-23; lanes16-31 K=8-15,24-31
        bfrag.h[0] = *(const v8bf*)(bp + 32 * dk);
        bfrag.h[1] = *(const v8bf*)(bp + 32 * dk + 16);
        acc0 = __builtin_amdgcn_wmma_f32_16x16x32_bf16(
                 false, afrag0[dk].v, false, bfrag.v, (short)0, acc0, false, false);
        acc1 = __builtin_amdgcn_wmma_f32_16x16x32_bf16(
                 false, afrag1[dk].v, false, bfrag.v, (short)0, acc1, false, false);
      }
      const float de = esqS[codeLocal];
      const int   gc = c * CCHUNK + codeLocal;
#pragma unroll
      for (int i = 0; i < 8; ++i) {
        float d0 = de - 2.0f * acc0[i];          // ||e||^2 - 2 z.e  (argmin-equivalent)
        float d1 = de - 2.0f * acc1[i];
        if (d0 < minv0[i]) { minv0[i] = d0; mini0[i] = gc; }
        if (d1 < minv1[i]) { minv1[i] = d1; mini1[i] = gc; }
      }
    }
  }

  // argmin across the 16 lanes of each half (columns), rows stay lane-local
#pragma unroll
  for (int off = 1; off < 16; off <<= 1) {
#pragma unroll
    for (int i = 0; i < 8; ++i) {
      float ov0 = __shfl_xor(minv0[i], off, 32);
      int   oi0 = __shfl_xor(mini0[i], off, 32);
      if (ov0 < minv0[i] || (ov0 == minv0[i] && oi0 < mini0[i])) { minv0[i] = ov0; mini0[i] = oi0; }
      float ov1 = __shfl_xor(minv1[i], off, 32);
      int   oi1 = __shfl_xor(mini1[i], off, 32);
      if (ov1 < minv1[i] || (ov1 == minv1[i] && oi1 < mini1[i])) { minv1[i] = ov1; mini1[i] = oi1; }
    }
  }

  // lane0 -> rows +0..7, lane16 -> rows +8..15 (per 16-row tile)
  if (lo16 == 0) {
    const int mrow0 = rowBase + hi * 8;          // tile 0
    const int mrow1 = rowBase + 16 + hi * 8;     // tile 1
    float lsum = 0.f;
#pragma unroll
    for (int i = 0; i < 8; ++i) {
      idxS[mrow0 + i] = mini0[i];
      idxS[mrow1 + i] = mini1[i];
      out[(size_t)NZ + (size_t)(n0 + mrow0 + i)] = (float)mini0[i];
      out[(size_t)NZ + (size_t)(n0 + mrow1 + i)] = (float)mini1[i];
      lsum += zsqS[mrow0 + i] + minv0[i];        // ||z||^2 + (||e||^2 - 2 z.e)
      lsum += zsqS[mrow1 + i] + minv1[i];
    }
    atomicAdd(&lossS[0], lsum);
  }
  __syncthreads();

  // epilogue: z_q = codebook[idx] (exact f32), NCHW stores coalesced over hw
  {
    const int r = tid;
    const int q = idxS[r];
    const float* crow = cb + (size_t)q * DDIM;
    float* oBase = out + (size_t)b * ((size_t)DDIM * HW) + hw0;
#pragma unroll 4
    for (int d = 0; d < DDIM; ++d) {
      oBase[(size_t)d * HW + r] = crow[d];
    }
  }
  if (tid == 0) atomicAdd(out + LOSS_OFF, lossS[0] * INV_N);
}

extern "C" void kernel_launch(void* const* d_in, const int* in_sizes, int n_in,
                              void* d_out, int out_size, void* d_ws, size_t ws_size,
                              hipStream_t stream) {
  const float* z  = (const float*)d_in[0];   // (32,256,64,64) f32
  const float* cb = (const float*)d_in[1];   // (1024,256) f32
  float* out = (float*)d_out;
  float* esq = (float*)d_ws;                 // 1024 f32 scratch

  vq_prep<<<4, 256, 0, stream>>>(cb, esq, out + LOSS_OFF);

  size_t smem = (size_t)(MTILE * PITCH + CCHUNK * PITCH) * sizeof(unsigned short)
              + (CCHUNK + MTILE) * sizeof(float)
              + MTILE * sizeof(int) + sizeof(float);
  vq_main<<<NIDX / MTILE, 256, smem, stream>>>(z, cb, esq, out);
}
